// DualAnchorContrastiveLoss_48936857371357
// MI455X (gfx1250) — matrix operand order repaired and checked by reading, not compiled, not run on hardware
//
#include <hip/hip_runtime.h>
#include <hip/hip_bf16.h>

typedef __attribute__((ext_vector_type(16))) _Float16 v16h;
typedef __attribute__((ext_vector_type(8)))  _Float16 v8h;
typedef __attribute__((ext_vector_type(8)))  float    v8f;
typedef __attribute__((ext_vector_type(4)))  float    f4;

// Global (addrspace 1) views so staging lowers to global_load_* (LOADcnt only),
// not flat_load_* (which couples LOADcnt+DScnt with the LDS pipeline).
typedef __attribute__((address_space(1))) const f4    gc_f4;
typedef __attribute__((address_space(1))) const float gc_f32;
typedef __attribute__((address_space(1)))       float g_f32;

static __device__ __forceinline__ const gc_f4*  as_g4(const float* p)  { return (const gc_f4*)(unsigned long long)p; }
static __device__ __forceinline__ const gc_f32* as_g (const float* p)  { return (const gc_f32*)(unsigned long long)p; }
static __device__ __forceinline__       g_f32*  as_gw(float* p)        { return (g_f32*)(unsigned long long)p; }

#define BROWS 4096
#define DIMK  1024
#define BM 128
#define BN 128
#define BK 32
#define LDS_STRIDE 40            // halfs; padded rows (80B) keep 16B alignment for b128 LDS ops
#define NPANEL (BROWS / BM)      // 32
#define NCHUNK (BROWS / BN)      // 32
#define NKT    (DIMK / BK)       // 32
#define NDIR  10
#define NPAIR 5
#define DIAG_BLKS (BROWS / 8)    // 512

// ---------------------------------------------------------------------------
// One block = one 128-row panel of one directed GEMM. Streams all 4096 columns
// with an online softmax (max, sumexp) per row held in LDS. Software-pipelined:
// global loads for K-tile kt+1 are in flight while WMMAs for tile kt run.
// ---------------------------------------------------------------------------
__global__ __launch_bounds__(256)
void lse_gemm_kernel(const float* __restrict__ t_emb, const float* __restrict__ i_emb,
                     const float* __restrict__ a_emb, const float* __restrict__ d_emb,
                     const float* __restrict__ scale_ptr, float* __restrict__ lse_part)
{
    __shared__ _Float16 As[2][BM * LDS_STRIDE];
    __shared__ _Float16 Bs[2][BN * LDS_STRIDE];
    __shared__ float rmax[2][BM];
    __shared__ float rsum[2][BM];
    __shared__ float redbuf[BM];

    const int tid    = threadIdx.x;
    const int lane   = tid & 31;
    const int wave   = tid >> 5;
    const int wave_m = wave & 3;   // 0..3 -> 32-row slice
    const int wave_n = wave >> 2;  // 0..1 -> 64-col slice
    const int dir    = blockIdx.y;
    const int panel  = blockIdx.x;

    const float* ptrs[4] = { t_emb, i_emb, a_emb, d_emb };
    const int dA[NDIR] = {2,0,2,1,3,0,3,1,0,1};
    const int dB[NDIR] = {0,2,1,2,0,3,1,3,1,0};
    const float* f1 = ptrs[dA[dir]];
    const float* f2 = ptrs[dB[dir]];
    const float sc = scale_ptr[0];

    for (int r = tid; r < BM; r += 256) {
        rmax[0][r] = -3.0e38f; rmax[1][r] = -3.0e38f;
        rsum[0][r] = 0.0f;     rsum[1][r] = 0.0f;
    }

    const int m0   = panel * BM;
    const int srow = tid >> 1;          // staging: each thread moves 16 f32 of A and of B
    const int skp  = (tid & 1) * 16;

    const int l15   = lane & 15;
    const int lhalf = lane >> 4;
    const int akoff0 = lhalf * 8;       // A frag: interleaved 16-bit layout (K 0-7/16-23 | 8-15/24-31)
    const int akoff1 = 16 + lhalf * 8;
    const int bkoff  = lhalf * 8;       // B frag: interleaved 16-bit layout (K 0-7/16-23 | 8-15/24-31)

    f4 ra[4], rb[4];                    // staging registers (global->regs leg of the pipeline)

    auto load_g = [&](int n0, int k0) {
        const gc_f4* ga = as_g4(f1 + (size_t)(m0 + srow) * DIMK + k0 + skp);
        const gc_f4* gb = as_g4(f2 + (size_t)(n0 + srow) * DIMK + k0 + skp);
        ra[0] = ga[0]; ra[1] = ga[1]; ra[2] = ga[2]; ra[3] = ga[3];
        rb[0] = gb[0]; rb[1] = gb[1]; rb[2] = gb[2]; rb[3] = gb[3];
    };
    auto store_lds = [&](int buf) {
        v8h a0, a1, b0, b1;
        #pragma unroll
        for (int j = 0; j < 8; ++j) {
            a0[j] = (_Float16)ra[j >> 2][j & 3];
            a1[j] = (_Float16)ra[2 + (j >> 2)][j & 3];
            b0[j] = (_Float16)rb[j >> 2][j & 3];
            b1[j] = (_Float16)rb[2 + (j >> 2)][j & 3];
        }
        *reinterpret_cast<v8h*>(&As[buf][srow * LDS_STRIDE + skp    ]) = a0;
        *reinterpret_cast<v8h*>(&As[buf][srow * LDS_STRIDE + skp + 8]) = a1;
        *reinterpret_cast<v8h*>(&Bs[buf][srow * LDS_STRIDE + skp    ]) = b0;
        *reinterpret_cast<v8h*>(&Bs[buf][srow * LDS_STRIDE + skp + 8]) = b1;
    };

    for (int nc = 0; nc < NCHUNK; ++nc) {
        const int n0 = nc * BN;

        v8f acc[2][4];
        #pragma unroll
        for (int a = 0; a < 2; ++a)
            #pragma unroll
            for (int b = 0; b < 4; ++b)
                #pragma unroll
                for (int e = 0; e < 8; ++e) acc[a][b][e] = 0.0f;

        // prime the pipeline: K-tile 0 into buffer 0
        load_g(n0, 0);
        __syncthreads();                 // previous chunk done reading LDS
        store_lds(0);

        for (int kt = 0; kt < NKT; ++kt) {
            const int cur = kt & 1;
            __syncthreads();             // buffer `cur` staged, buffer cur^1 free

            if (kt + 1 < NKT) load_g(n0, (kt + 1) * BK);   // loads in flight during WMMA
            if (kt + 2 < NKT) {                             // warm L2 for the tile after that
                __builtin_prefetch(f1 + (size_t)(m0 + srow) * DIMK + (kt + 2) * BK + skp, 0, 3);
                __builtin_prefetch(f2 + (size_t)(n0 + srow) * DIMK + (kt + 2) * BK + skp, 0, 3);
            }

            // ---- fragment loads + WMMA on buffer `cur` ----
            v16h afrag[2], bfrag[4];
            #pragma unroll
            for (int tm = 0; tm < 2; ++tm) {
                const int row = wave_m * 32 + tm * 16 + l15;
                v8h lo = *reinterpret_cast<const v8h*>(&As[cur][row * LDS_STRIDE + akoff0]);
                v8h hi = *reinterpret_cast<const v8h*>(&As[cur][row * LDS_STRIDE + akoff1]);
                union { v16h v; v8h h[2]; } u; u.h[0] = lo; u.h[1] = hi;
                afrag[tm] = u.v;
            }
            #pragma unroll
            for (int tn = 0; tn < 4; ++tn) {
                const int col = wave_n * 64 + tn * 16 + l15;
                v8h lo = *reinterpret_cast<const v8h*>(&Bs[cur][col * LDS_STRIDE + bkoff    ]);
                v8h hi = *reinterpret_cast<const v8h*>(&Bs[cur][col * LDS_STRIDE + bkoff + 16]);
                union { v16h v; v8h h[2]; } u; u.h[0] = lo; u.h[1] = hi;
                bfrag[tn] = u.v;
            }
            #pragma unroll
            for (int tm = 0; tm < 2; ++tm)
                #pragma unroll
                for (int tn = 0; tn < 4; ++tn)
                    acc[tm][tn] = __builtin_amdgcn_wmma_f32_16x16x32_f16(
                        false, afrag[tm], false, bfrag[tn],
                        (short)0, acc[tm][tn], false, false);

            // drain the in-flight loads into the other buffer (after the WMMAs)
            if (kt + 1 < NKT) store_lds(cur ^ 1);
        }

        // ---- online row-softmax update for this 128-column chunk ----
        // C frag layout: element r -> row (lane<16 ? r : r+8) of the 16x16 tile, col = lane&15.
        #pragma unroll
        for (int tm = 0; tm < 2; ++tm) {
            #pragma unroll
            for (int r = 0; r < 8; ++r) {
                float v0 = sc * acc[tm][0][r];
                float v1 = sc * acc[tm][1][r];
                float v2 = sc * acc[tm][2][r];
                float v3 = sc * acc[tm][3][r];
                float mloc = fmaxf(fmaxf(v0, v1), fmaxf(v2, v3));
                #pragma unroll
                for (int off = 1; off < 16; off <<= 1)
                    mloc = fmaxf(mloc, __shfl_xor(mloc, off, 32));
                float sl = expf(v0 - mloc) + expf(v1 - mloc) + expf(v2 - mloc) + expf(v3 - mloc);
                #pragma unroll
                for (int off = 1; off < 16; off <<= 1)
                    sl += __shfl_xor(sl, off, 32);
                if (l15 == 0) {
                    const int rrow = wave_m * 32 + tm * 16 + r + lhalf * 8;
                    float om = rmax[wave_n][rrow];
                    float nm = fmaxf(om, mloc);
                    rsum[wave_n][rrow] = rsum[wave_n][rrow] * expf(om - nm) + sl * expf(mloc - nm);
                    rmax[wave_n][rrow] = nm;
                }
            }
        }
    }

    __syncthreads();
    if (tid < BM) {
        float ma = rmax[0][tid], mb = rmax[1][tid];
        float M  = fmaxf(ma, mb);
        float ss = rsum[0][tid] * expf(ma - M) + rsum[1][tid] * expf(mb - M);
        redbuf[tid] = M + logf(ss);
    }
    __syncthreads();
    if (tid == 0) {
        float a = 0.0f;
        for (int r = 0; r < BM; ++r) a += redbuf[r];
        as_gw(lse_part)[dir * NPANEL + panel] = a;   // deterministic: one writer per slot
    }
}

// ---------------------------------------------------------------------------
// Diagonal dot products: one wave per row, 5 pairs. F32 exact.
// ---------------------------------------------------------------------------
__global__ __launch_bounds__(256)
void diag_kernel(const float* __restrict__ t_emb, const float* __restrict__ i_emb,
                 const float* __restrict__ a_emb, const float* __restrict__ d_emb,
                 float* __restrict__ diag_part)
{
    __shared__ float part[8];
    const int lane = threadIdx.x & 31;
    const int wave = threadIdx.x >> 5;
    const int pair = blockIdx.y;
    const float* ptrs[4] = { t_emb, i_emb, a_emb, d_emb };
    const int pA[NPAIR] = {2,2,3,3,0};
    const int pB[NPAIR] = {0,1,0,1,1};
    const size_t rowoff = (size_t)(blockIdx.x * 8 + wave) * DIMK;
    const gc_f32* r1 = as_g(ptrs[pA[pair]] + rowoff);
    const gc_f32* r2 = as_g(ptrs[pB[pair]] + rowoff);
    float dot = 0.0f;
    for (int k = lane; k < DIMK; k += 32) dot = fmaf(r1[k], r2[k], dot);
    #pragma unroll
    for (int off = 1; off < 32; off <<= 1) dot += __shfl_xor(dot, off, 32);
    if (lane == 0) part[wave] = dot;
    __syncthreads();
    if (threadIdx.x == 0) {
        float s = 0.0f;
        for (int w = 0; w < 8; ++w) s += part[w];
        as_gw(diag_part)[pair * DIAG_BLKS + blockIdx.x] = s;
    }
}

// ---------------------------------------------------------------------------
// Final deterministic scalar reduction:
//   loss = 0.5/B * sum(all row-LSEs over 10 directions) - s/B * sum(pair diagonals)
// ---------------------------------------------------------------------------
__global__ void final_kernel(const float* __restrict__ lse_part,
                             const float* __restrict__ diag_part,
                             const float* __restrict__ scale_ptr,
                             float* __restrict__ out)
{
    if (threadIdx.x == 0 && blockIdx.x == 0) {
        double accL = 0.0, accD = 0.0;
        for (int i = 0; i < NDIR * NPANEL; ++i)     accL += (double)lse_part[i];
        for (int i = 0; i < NPAIR * DIAG_BLKS; ++i) accD += (double)diag_part[i];
        const double s = (double)scale_ptr[0];
        out[0] = (float)(0.5 * accL / (double)BROWS - s * accD / (double)BROWS);
    }
}

extern "C" void kernel_launch(void* const* d_in, const int* in_sizes, int n_in,
                              void* d_out, int out_size, void* d_ws, size_t ws_size,
                              hipStream_t stream) {
    const float* t_emb = (const float*)d_in[0];
    const float* i_emb = (const float*)d_in[1];
    const float* a_emb = (const float*)d_in[2];
    const float* d_emb = (const float*)d_in[3];
    const float* scale = (const float*)d_in[4];
    float* ws        = (float*)d_ws;
    float* lse_part  = ws;                    // 10*32   = 320 floats
    float* diag_part = ws + NDIR * NPANEL;    // 5*512   = 2560 floats  (~11.5 KB total)

    dim3 g1(NPANEL, NDIR);       // 32 panels x 10 directed GEMMs = 320 blocks
    lse_gemm_kernel<<<g1, 256, 0, stream>>>(t_emb, i_emb, a_emb, d_emb, scale, lse_part);

    dim3 g2(DIAG_BLKS, NPAIR);
    diag_kernel<<<g2, 256, 0, stream>>>(t_emb, i_emb, a_emb, d_emb, diag_part);

    final_kernel<<<1, 32, 0, stream>>>(lse_part, diag_part, scale, (float*)d_out);
}
